// BiQLSTM_35880156791527
// MI455X (gfx1250) — compile-verified, matrix operand order checked by reading, not run on hardware
//
#include <hip/hip_runtime.h>
#include <hip/hip_bf16.h>

// ---------------------------------------------------------------------------
// BiQLSTM for MI455X (gfx1250, wave32).
//
// Roofline: ~1.7 GFLOP total, ~260 MB min traffic (~11us @ 23.3 TB/s).
// The T=1024 recurrence is the critical path, so:
//   (1) the non-recurrent half of the input projection (x @ Wx^T for BOTH
//       directions, feature-reversal folded into bw weights) is hoisted into
//       one [131072 x 128] x [128 x 16] fp32 WMMA GEMM (v_wmma_f32_16x16x4_f32),
//   (2) the recurrence runs as 256 independent (dir,batch) sequences, one
//       4-wave workgroup each: gate-per-wave butterfly reduction for W_h*h,
//       16-lane VQC stage, lane-per-h output stage, 3 barriers/step.
// ---------------------------------------------------------------------------

typedef float  v2f   __attribute__((ext_vector_type(2)));
typedef float  v8f   __attribute__((ext_vector_type(8)));
typedef __bf16 v16bf __attribute__((ext_vector_type(16)));

#define T_LEN 1024
#define BATCH 128

// ---------------- kernel 0: pack x-side weights --------------------------
// wxT[16][128]: rows 0..3 = fw_w_in[k][128+f], rows 4..7 = bw_w_in[k][128+127-f]
// (folds the bw feature reversal of x into the weights), rows 8..15 = 0.
__global__ void wxT_setup_kernel(const float* __restrict__ fw_w_in,
                                 const float* __restrict__ bw_w_in,
                                 float* __restrict__ wxT) {
    int f = threadIdx.x;  // 0..127
#pragma unroll
    for (int n = 0; n < 16; ++n) {
        float v = 0.0f;
        if (n < 4)      v = fw_w_in[n * 256 + 128 + f];
        else if (n < 8) v = bw_w_in[(n - 4) * 256 + 128 + (127 - f)];
        wxT[n * 128 + f] = v;
    }
}

// ---------------- kernel 1: yx = x @ wxT^T via WMMA ----------------------
// M = B*T = 131072 rows of x (K=128), N=16 (cols 0..7 real: 4 fw + 4 bw).
// One wave per 16-row tile. fp32 WMMA 16x16x4, 32 K-steps, accumulate in v8f.
// A layout (32b 16x4): lane = M (l&15), VGPR v: K = (l>>4)*2 + v  -> float2.
// B layout mirrors A with lane = N. C/D: VGPR r, lane l -> M=r+(l>>4)*8, N=l&15.
__global__ __launch_bounds__(256) void yx_gemm_kernel(
    const float* __restrict__ x, const float* __restrict__ wxT,
    float* __restrict__ yx) {
    const int lane   = threadIdx.x & 31;
    const int wave   = threadIdx.x >> 5;
    const int tile   = blockIdx.x * 8 + wave;      // 0..8191
    const int rowb   = tile * 16;
    const int mloc   = lane & 15;
    const int hi     = lane >> 4;                  // 0 or 1

    v8f c = {};

#if __has_builtin(__builtin_amdgcn_wmma_f32_16x16x4_f32)
    const float* xa = x   + (rowb + mloc) * 128 + hi * 2;
    const float* wb = wxT + mloc * 128 + hi * 2;
#pragma unroll
    for (int s = 0; s < 32; ++s) {
        v2f a = *(const v2f*)(xa + s * 4);
        v2f b = *(const v2f*)(wb + s * 4);
        c = __builtin_amdgcn_wmma_f32_16x16x4_f32(
                false, a, false, b, (short)0, c, false, false);
    }
#else
    // Fallback: probe-confirmed bf16 path, K=128 in 4 steps of 32.
    // 16-bit A 16x32 layout: lane half selects K-base offset 8; per lane
    // halves are K = kb+off+{0..7} then K = kb+off+16+{0..7}.
    const int off = hi * 8;
    const float* xa = x   + (rowb + mloc) * 128 + off;
    const float* wb = wxT + mloc * 128 + off;
#pragma unroll
    for (int s = 0; s < 4; ++s) {
        const float* pa = xa + s * 32;
        const float* pb = wb + s * 32;
        v16bf a, b;
#pragma unroll
        for (int i = 0; i < 8; ++i) {
            a[i]     = (__bf16)pa[i];
            a[8 + i] = (__bf16)pa[16 + i];
            b[i]     = (__bf16)pb[i];
            b[8 + i] = (__bf16)pb[16 + i];
        }
        c = __builtin_amdgcn_wmma_f32_16x16x32_bf16(
                false, a, false, b, (short)0, c, false, false);
    }
#endif

    if (mloc < 8) {                                 // only 8 real columns
        const int mhi = hi * 8;
#pragma unroll
        for (int r = 0; r < 8; ++r)
            yx[(rowb + mhi + r) * 8 + mloc] = c[r];
    }
}

// ---------------- fast activations ---------------------------------------
__device__ __forceinline__ float fast_sigmoid(float x) {
    return __builtin_amdgcn_rcpf(1.0f + __expf(-x));
}
__device__ __forceinline__ float fast_tanh(float x) {
    float e = __expf(2.0f * x);
    return 1.0f - 2.0f * __builtin_amdgcn_rcpf(e + 1.0f);
}

// ---------------- kernel 2: recurrence ------------------------------------
// grid = 256 blocks (blockIdx < 128: fw batch b; >= 128: bw batch b-128),
// block = 128 threads (4 waves). Lane tid owns h-index tid.
__global__ __launch_bounds__(128) void qlstm_rec_kernel(
    const float* __restrict__ yx,
    const float* fw_w_in, const float* fw_b_in, const float* fw_w_vqc,
    const float* fw_b_vqc, const float* fw_w_out, const float* fw_b_out,
    const float* bw_w_in, const float* bw_b_in, const float* bw_w_vqc,
    const float* bw_b_vqc, const float* bw_w_out, const float* bw_b_out,
    float* __restrict__ out) {
    __shared__ __align__(16) float h_lds[128];
    __shared__ __align__(16) float hsum_lds[4];
    __shared__ __align__(16) float q_lds[16];

    const int tid = threadIdx.x;
    const int dir = blockIdx.x >> 7;
    const int b   = blockIdx.x & 127;

    const float* w_in  = dir ? bw_w_in  : fw_w_in;
    const float* b_in  = dir ? bw_b_in  : fw_b_in;
    const float* w_vqc = dir ? bw_w_vqc : fw_w_vqc;
    const float* b_vqc = dir ? bw_b_vqc : fw_b_vqc;
    const float* w_out = dir ? bw_w_out : fw_w_out;
    const float* b_out = dir ? bw_b_out : fw_b_out;

    // stage A preload: wave = gate, lane dots 4 h's against W_h slice
    const int gate = tid >> 5, l = tid & 31;
    const float4 wh4 = *(const float4*)(w_in + gate * 256 + 4 * l);

    // stage C preload (lanes 0..15): one VQC output (k = tid>>2, p = tid&3)
    const int ck = (tid >> 2) & 3, cp = tid & 3;
    const float wv0 = w_vqc[ck * 16 + 0 + cp];
    const float wv1 = w_vqc[ck * 16 + 4 + cp];
    const float wv2 = w_vqc[ck * 16 + 8 + cp];
    const float wv3 = w_vqc[ck * 16 + 12 + cp];
    const float bvq = b_vqc[tid & 15];
    const float4 bi4 = *(const float4*)(b_in);

    // stage E preload: this lane's w_out row + bias
    const float4 wo4 = *(const float4*)(w_out + tid * 4);
    const float  bo  = b_out[tid];

    const float* up = yx + (long)b * T_LEN * 8 + dir * 4;
    float* outseq   = out + (long)blockIdx.x * T_LEN * 128;
    const int col   = dir ? (127 - tid) : tid;   // bw output feature reversal

    float c  = 0.0f;
    float hn = 0.0f;
    h_lds[tid] = 0.0f;
    __syncthreads();

    for (int s = 0; s < T_LEN; ++s) {
        // ---- stage A: per-gate W_h*h partial + in-wave butterfly ----
        float4 h4 = *(const float4*)&h_lds[4 * l];
        float part = h4.x * wh4.x + h4.y * wh4.y + h4.z * wh4.z + h4.w * wh4.w;
        part += __shfl_xor(part, 16);
        part += __shfl_xor(part, 8);
        part += __shfl_xor(part, 4);
        part += __shfl_xor(part, 2);
        part += __shfl_xor(part, 1);
        if (l == 0) hsum_lds[gate] = part;
        __syncthreads();

        // ---- stage C: 16 lanes compute the 4x4 VQC block ----
        if (tid < 16) {
            const int tu = dir ? (T_LEN - 1 - s) : s;   // bw time reversal
            float4 u4 = *(const float4*)(up + tu * 8);
            float y0 = hsum_lds[0] + u4.x + bi4.x;
            float y1 = hsum_lds[1] + u4.y + bi4.y;
            float y2 = hsum_lds[2] + u4.z + bi4.z;
            float y3 = hsum_lds[3] + u4.w + bi4.w;
            float acc = bvq + y0 * wv0 + y1 * wv1 + y2 * wv2 + y3 * wv3;
            q_lds[tid] = fast_tanh(acc);
        }
        __syncthreads();

        // ---- stage E: z = q @ w_out^T for this lane's h, gate + update ----
        float4 q0 = *(const float4*)&q_lds[0];
        float4 q1 = *(const float4*)&q_lds[4];
        float4 q2 = *(const float4*)&q_lds[8];
        float4 q3 = *(const float4*)&q_lds[12];
        float z0 = bo + q0.x * wo4.x + q0.y * wo4.y + q0.z * wo4.z + q0.w * wo4.w;
        float z1 = bo + q1.x * wo4.x + q1.y * wo4.y + q1.z * wo4.z + q1.w * wo4.w;
        float z2 = bo + q2.x * wo4.x + q2.y * wo4.y + q2.z * wo4.z + q2.w * wo4.w;
        float z3 = bo + q3.x * wo4.x + q3.y * wo4.y + q3.z * wo4.z + q3.w * wo4.w;
        float ig = fast_sigmoid(z0);
        float fg = fast_sigmoid(z1);
        float gg = fast_tanh(z2);
        float og = fast_sigmoid(z3);
        c  = fg * c + ig * gg;
        hn = og * fast_tanh(c);
        h_lds[tid] = hn;
        const int t_out = dir ? (T_LEN - 1 - s) : s;
        outseq[t_out * 128 + col] = hn;                 // coalesced 512B/step
        __syncthreads();
    }

    // final states: [h_fw, c_fw, h_bw, c_bw] after the 256*1024*128 sequence
    const long SBASE = 256L * T_LEN * 128;
    float* fin = out + SBASE + (long)dir * 2 * 16384 + b * 128 + tid;
    fin[0]     = hn;
    fin[16384] = c;
}

// ---------------- launch ---------------------------------------------------
extern "C" void kernel_launch(void* const* d_in, const int* in_sizes, int n_in,
                              void* d_out, int out_size, void* d_ws, size_t ws_size,
                              hipStream_t stream) {
    (void)in_sizes; (void)n_in; (void)out_size; (void)ws_size;
    const float* x         = (const float*)d_in[0];
    const float* fw_w_in   = (const float*)d_in[1];
    const float* fw_b_in   = (const float*)d_in[2];
    const float* fw_w_vqc  = (const float*)d_in[3];
    const float* fw_b_vqc  = (const float*)d_in[4];
    const float* fw_w_out  = (const float*)d_in[5];
    const float* fw_b_out  = (const float*)d_in[6];
    const float* bw_w_in   = (const float*)d_in[7];
    const float* bw_b_in   = (const float*)d_in[8];
    const float* bw_w_vqc  = (const float*)d_in[9];
    const float* bw_b_vqc  = (const float*)d_in[10];
    const float* bw_w_out  = (const float*)d_in[11];
    const float* bw_b_out  = (const float*)d_in[12];

    float* wxT = (float*)d_ws;            // 16*128 floats = 8 KB
    float* yx  = (float*)d_ws + 2048;     // 131072*8 floats = 4 MB

    wxT_setup_kernel<<<1, 128, 0, stream>>>(fw_w_in, bw_w_in, wxT);
    yx_gemm_kernel<<<1024, 256, 0, stream>>>(x, wxT, yx);
    qlstm_rec_kernel<<<256, 128, 0, stream>>>(
        yx,
        fw_w_in, fw_b_in, fw_w_vqc, fw_b_vqc, fw_w_out, fw_b_out,
        bw_w_in, bw_b_in, bw_w_vqc, bw_b_vqc, bw_w_out, bw_b_out,
        (float*)d_out);
}
